// GATAutoEncoder_80522046865471
// MI455X (gfx1250) — matrix operand order repaired and checked by reading, not compiled, not run on hardware
//
#include <hip/hip_runtime.h>
#include <hip/hip_bf16.h>
#include <math.h>

// ---------------- problem constants (match reference setup_inputs) ----------
#define NNODES 30000
#define NEDGES 480000
#define NTOT   (NEDGES + NNODES)   // edges + self loops
#define DIN    1000
#define NHEAD  8
#define FH     32
#define HF     256                 // NHEAD*FH
#define DEMB   64
#define KPAD1  1024                // DIN padded to mult of 32
#define NPADD2 1024                // DIN padded so 64 column tiles divide by NT=8

typedef __attribute__((ext_vector_type(16))) __bf16 v16bf;
typedef __attribute__((ext_vector_type(8)))  float  v8f;

// ---- order-preserving float <-> uint encoding for atomic segment-max ------
__device__ __forceinline__ unsigned fenc(float f) {
    unsigned u = __float_as_uint(f);
    return (u & 0x80000000u) ? ~u : (u | 0x80000000u);
}
__device__ __forceinline__ float fdec(unsigned u) {
    u = (u & 0x80000000u) ? (u & 0x7FFFFFFFu) : ~u;
    return __uint_as_float(u);
}

// native no-return f32 atomic add (global_atomic_add_f32) on AMD
__device__ __forceinline__ void fatomic_add(float* p, float v) {
    unsafeAtomicAdd(p, v);
}

// ---------------------------------------------------------------------------
// Weight convert + transpose + pad:  W[K,Ncols] f32 -> out[Npad,Kpad] bf16
// ---------------------------------------------------------------------------
__global__ void gat_ae_convert_wT(const float* __restrict__ W, __bf16* __restrict__ out,
                                  int K, int Ncols, int Kpad, int Npad) {
    int t = blockIdx.x * blockDim.x + threadIdx.x;
    if (t >= Npad * Kpad) return;
    int n = t / Kpad, k = t - n * Kpad;
    float v = (n < Ncols && k < K) ? W[(size_t)k * Ncols + n] : 0.0f;
    out[t] = (__bf16)v;
}

// ---------------------------------------------------------------------------
// Per-launch init: zero aggregation buffer, init segment max/denominator
// ---------------------------------------------------------------------------
__global__ void gat_ae_init(float* __restrict__ outgat, unsigned* __restrict__ mEnc,
                            float* __restrict__ denom) {
    int t = blockIdx.x * blockDim.x + threadIdx.x;
    if (t < NNODES * HF) outgat[t] = 0.0f;
    if (t < NNODES * NHEAD) { mEnc[t] = 0x007FFFFFu /* fenc(-inf) */; denom[t] = 0.0f; }
}

// ---------------------------------------------------------------------------
// GEMM: C[M,Ncols] = A[M,K](f32) * B  (+bias, optional relu)
// B pre-transposed/padded: Bt[Npad,Kpad] bf16, row n = original column n.
// One wave per 16x(16*NT) strip: one A fragment feeds NT WMMAs per K-step.
// Branch-free main loop (K & ~31) + single guarded tail iteration.
// ---------------------------------------------------------------------------
template<int NT>
__global__ __launch_bounds__(32)
void gat_ae_gemm_wmma(const float* __restrict__ A, int lda,
                      const __bf16* __restrict__ Bt, int Kpad, int K,
                      const float* __restrict__ bias, int Ncols,
                      float* __restrict__ C, int ldc, int doRelu) {
    const int lane = threadIdx.x;        // 0..31
    const int m0   = blockIdx.x * 16;
    const int n0   = blockIdx.y * (16 * NT);
    const int hfId = lane >> 4;          // 0/1
    const int l16  = lane & 15;
    const int kbA  = hfId * 16;          // A: contiguous 16-K chunk per lane half
    const int kbB  = hfId * 16;          // B: contiguous 16-K chunk per lane half

    const float*  Arow = A  + (size_t)(m0 + l16) * lda;
    const __bf16* Brow = Bt + (size_t)(n0 + l16) * Kpad;   // tile j adds 16*j*Kpad

    v8f acc[NT];
#pragma unroll
    for (int j = 0; j < NT; ++j) acc[j] = (v8f){};

    const int kmain = K & ~31;
    for (int k = 0; k < kmain; k += 32) {
        // A fragment: 8+8 consecutive f32 -> bf16 (v_cvt_pk_bf16_f32)
        v8f x0 = *(const v8f*)(Arow + k + kbA);
        v8f x1 = *(const v8f*)(Arow + k + kbA + 8);
        __builtin_prefetch((const void*)(Arow + k + 32), 0, 3);  // global_prefetch_b8
        v16bf a;
#pragma unroll
        for (int i = 0; i < 8; ++i) { a[i] = (__bf16)x0[i]; a[8 + i] = (__bf16)x1[i]; }

        // issue all NT B-tile loads first, then the WMMA chain
        v16bf b[NT];
#pragma unroll
        for (int j = 0; j < NT; ++j)
            b[j] = *(const v16bf*)(Brow + (size_t)(16 * j) * Kpad + k + kbB);
#pragma unroll
        for (int j = 0; j < NT; ++j)
            acc[j] = __builtin_amdgcn_wmma_f32_16x16x32_bf16(
                         false, a, false, b[j], (short)0, acc[j], false, false);
    }
    if (kmain < Kpad) {                  // tail (GEMM1 only: K=1000, Kpad=1024)
        const int k = kmain;
        v16bf a;
#pragma unroll
        for (int i = 0; i < 8; ++i) {
            int k1 = k + kbA + i;
            int k2 = k + 8 + kbA + i;
            a[i]     = (k1 < K) ? (__bf16)Arow[k1] : (__bf16)0.0f;
            a[8 + i] = (k2 < K) ? (__bf16)Arow[k2] : (__bf16)0.0f;
        }
        v16bf b[NT];
#pragma unroll
        for (int j = 0; j < NT; ++j)
            b[j] = *(const v16bf*)(Brow + (size_t)(16 * j) * Kpad + k + kbB);
#pragma unroll
        for (int j = 0; j < NT; ++j)
            acc[j] = __builtin_amdgcn_wmma_f32_16x16x32_bf16(
                         false, a, false, b[j], (short)0, acc[j], false, false);
    }

#pragma unroll
    for (int j = 0; j < NT; ++j) {
        const int ccol = n0 + 16 * j + l16;
        if (ccol < Ncols) {
            float bv = bias ? bias[ccol] : 0.0f;
#pragma unroll
            for (int r = 0; r < 8; ++r) {
                int crow = m0 + r + hfId * 8;
                float v = acc[j][r] + bv;
                if (doRelu) v = fmaxf(v, 0.0f);
                C[(size_t)crow * ldc + ccol] = v;
            }
        }
    }
}

// ---------------------------------------------------------------------------
// a_s[n,h] = <h[n,h,:], att_src[h,:]> ; same for a_d
// ---------------------------------------------------------------------------
__global__ void gat_ae_attn_scores(const float* __restrict__ h,
                                   const float* __restrict__ att_s,
                                   const float* __restrict__ att_d,
                                   float* __restrict__ as_, float* __restrict__ ad_) {
    int t = blockIdx.x * blockDim.x + threadIdx.x;
    if (t >= NNODES * NHEAD) return;
    int n = t >> 3, hd = t & 7;
    const float* row = h + (size_t)n * HF + hd * FH;
    const float* vs  = att_s + hd * FH;
    const float* vd  = att_d + hd * FH;
    float s = 0.0f, d = 0.0f;
#pragma unroll 8
    for (int f = 0; f < FH; ++f) { float x = row[f]; s += x * vs[f]; d += x * vd[f]; }
    as_[t] = s; ad_[t] = d;
}

__device__ __forceinline__ void edge_ends(const int* __restrict__ ei, int e, int& s, int& d) {
    if (e < NEDGES) { s = ei[e]; d = ei[NEDGES + e]; }
    else            { s = d = e - NEDGES; }          // self loops
}

// ---------------------------------------------------------------------------
// logits + segment max (atomic umax on monotone encoding)
// ---------------------------------------------------------------------------
__global__ void gat_ae_edge_logits(const int* __restrict__ ei,
                                   const float* __restrict__ as_, const float* __restrict__ ad_,
                                   float* __restrict__ logits, unsigned* __restrict__ mEnc) {
    int t = blockIdx.x * blockDim.x + threadIdx.x;
    if (t >= NTOT * NHEAD) return;
    int e = t >> 3, hd = t & 7;
    int s, d; edge_ends(ei, e, s, d);
    float v = as_[s * NHEAD + hd] + ad_[d * NHEAD + hd];
    v = (v > 0.0f) ? v : 0.2f * v;                   // leaky relu
    logits[t] = v;
    atomicMax(mEnc + d * NHEAD + hd, fenc(v));
}

// ---------------------------------------------------------------------------
// p = exp(logit - m[dst]); denominator scatter-add; overwrite logits with p
// ---------------------------------------------------------------------------
__global__ void gat_ae_edge_exp(const int* __restrict__ ei,
                                float* __restrict__ logits,
                                const unsigned* __restrict__ mEnc,
                                float* __restrict__ denom) {
    int t = blockIdx.x * blockDim.x + threadIdx.x;
    if (t >= NTOT * NHEAD) return;
    int e = t >> 3, hd = t & 7;
    int s, d; edge_ends(ei, e, s, d);
    float p = __expf(logits[t] - fdec(mEnc[d * NHEAD + hd]));
    logits[t] = p;
    fatomic_add(denom + d * NHEAD + hd, p);
}

// ---------------------------------------------------------------------------
// message scatter: out[dst, c] += h[src, c] * p/denom    (block = edge)
// ---------------------------------------------------------------------------
__global__ __launch_bounds__(HF)
void gat_ae_edge_scatter(const int* __restrict__ ei,
                         const float* __restrict__ pbuf,
                         const float* __restrict__ denom,
                         const float* __restrict__ h,
                         float* __restrict__ outgat) {
    int e = blockIdx.x;
    int c = threadIdx.x;                 // 0..255
    int hd = c >> 5;
    int s, d; edge_ends(ei, e, s, d);
    float alpha = pbuf[e * NHEAD + hd] / denom[d * NHEAD + hd];
    float val = h[(size_t)s * HF + c] * alpha;
    fatomic_add(outgat + (size_t)d * HF + c, val);
}

__global__ void gat_ae_bias_relu(float* __restrict__ outgat, const float* __restrict__ bias) {
    int t = blockIdx.x * blockDim.x + threadIdx.x;
    if (t >= NNODES * HF) return;
    float v = outgat[t] + bias[t & (HF - 1)];
    outgat[t] = fmaxf(v, 0.0f);
}

// ---------------------------------------------------------------------------
// workspace layout (bytes, all 256B aligned)
// ---------------------------------------------------------------------------
#define OFF_H     ((size_t)0)                         // h = x@W        [N,256] f32
#define OFF_OUTG  (OFF_H    + (size_t)NNODES*HF*4)    // aggregated GAT [N,256] f32
#define OFF_AS    (OFF_OUTG + (size_t)NNODES*HF*4)    // a_src          [N,8]
#define OFF_AD    (OFF_AS   + (size_t)NNODES*NHEAD*4)
#define OFF_MENC  (OFF_AD   + (size_t)NNODES*NHEAD*4) // encoded segment max
#define OFF_DEN   (OFF_MENC + (size_t)NNODES*NHEAD*4)
#define OFF_P     (OFF_DEN  + (size_t)NNODES*NHEAD*4) // logits->p      [E+N,8]
#define OFF_D     (OFF_P    + (size_t)NTOT*NHEAD*4)   // decoder hidden [N,32]
#define OFF_WT1   (OFF_D    + (size_t)NNODES*FH*4)    // W^T bf16  [256,1024]
#define OFF_WT2   (OFF_WT1  + (size_t)HF*KPAD1*2)     // embW^T    [64,256]
#define OFF_WT3   (OFF_WT2  + (size_t)DEMB*HF*2)      // decW1^T   [32,64]
#define OFF_WT4   (OFF_WT3  + (size_t)FH*DEMB*2)      // decW2^T   [1024,32]

extern "C" void kernel_launch(void* const* d_in, const int* in_sizes, int n_in,
                              void* d_out, int out_size, void* d_ws, size_t ws_size,
                              hipStream_t stream) {
    const float* x        = (const float*)d_in[0];
    const int*   ei       = (const int*)  d_in[1];
    // d_in[2] edge_weight: unused by reference
    const float* W        = (const float*)d_in[3];
    const float* att_src  = (const float*)d_in[4];
    const float* att_dst  = (const float*)d_in[5];
    const float* bias_gat = (const float*)d_in[6];
    const float* emb_W    = (const float*)d_in[7];
    const float* emb_b    = (const float*)d_in[8];
    const float* dec_W1   = (const float*)d_in[9];
    const float* dec_b1   = (const float*)d_in[10];
    const float* dec_W2   = (const float*)d_in[11];
    const float* dec_b2   = (const float*)d_in[12];

    float* recon = (float*)d_out;                          // [30000,1000]
    float* z     = (float*)d_out + (size_t)NNODES * DIN;   // [30000,64]

    char* ws = (char*)d_ws;
    float*    hbuf   = (float*)   (ws + OFF_H);
    float*    outgat = (float*)   (ws + OFF_OUTG);
    float*    a_s    = (float*)   (ws + OFF_AS);
    float*    a_d    = (float*)   (ws + OFF_AD);
    unsigned* mEnc   = (unsigned*)(ws + OFF_MENC);
    float*    denom  = (float*)   (ws + OFF_DEN);
    float*    pbuf   = (float*)   (ws + OFF_P);
    float*    dbuf   = (float*)   (ws + OFF_D);
    __bf16*   wt1    = (__bf16*)  (ws + OFF_WT1);
    __bf16*   wt2    = (__bf16*)  (ws + OFF_WT2);
    __bf16*   wt3    = (__bf16*)  (ws + OFF_WT3);
    __bf16*   wt4    = (__bf16*)  (ws + OFF_WT4);

    // --- weight convert/transpose/pad (bf16) -------------------------------
    gat_ae_convert_wT<<<(HF * KPAD1 + 255) / 256, 256, 0, stream>>>(W,      wt1, DIN,  HF,   KPAD1, HF);
    gat_ae_convert_wT<<<(DEMB * HF  + 255) / 256, 256, 0, stream>>>(emb_W,  wt2, HF,   DEMB, HF,    DEMB);
    gat_ae_convert_wT<<<(FH * DEMB  + 255) / 256, 256, 0, stream>>>(dec_W1, wt3, DEMB, FH,   DEMB,  FH);
    gat_ae_convert_wT<<<(NPADD2 * FH + 255) / 256, 256, 0, stream>>>(dec_W2, wt4, FH,  DIN,  FH,    NPADD2);

    gat_ae_init<<<(NNODES * HF + 255) / 256, 256, 0, stream>>>(outgat, mEnc, denom);

    // --- GEMM1: h = x @ W   [30000,1000]x[1000,256] (WMMA bf16) ------------
    gat_ae_gemm_wmma<8><<<dim3(NNODES / 16, HF / 128), 32, 0, stream>>>(
        x, DIN, wt1, KPAD1, DIN, nullptr, HF, hbuf, HF, 0);

    // --- attention / segment softmax / scatter aggregation -----------------
    gat_ae_attn_scores<<<(NNODES * NHEAD + 255) / 256, 256, 0, stream>>>(
        hbuf, att_src, att_dst, a_s, a_d);
    gat_ae_edge_logits<<<(NTOT * NHEAD + 255) / 256, 256, 0, stream>>>(
        ei, a_s, a_d, pbuf, mEnc);
    gat_ae_edge_exp<<<(NTOT * NHEAD + 255) / 256, 256, 0, stream>>>(
        ei, pbuf, mEnc, denom);
    gat_ae_edge_scatter<<<NTOT, HF, 0, stream>>>(ei, pbuf, denom, hbuf, outgat);
    gat_ae_bias_relu<<<(NNODES * HF + 255) / 256, 256, 0, stream>>>(outgat, bias_gat);

    // --- z = gat @ emb_W + emb_b   [30000,256]x[256,64] --------------------
    gat_ae_gemm_wmma<4><<<dim3(NNODES / 16, 1), 32, 0, stream>>>(
        outgat, HF, wt2, HF, HF, emb_b, DEMB, z, DEMB, 0);

    // --- d = relu(z @ dec_W1 + b1)  [30000,64]x[64,32] ---------------------
    gat_ae_gemm_wmma<2><<<dim3(NNODES / 16, 1), 32, 0, stream>>>(
        z, DEMB, wt3, DEMB, DEMB, dec_b1, FH, dbuf, FH, 1);

    // --- recon = d @ dec_W2 + b2   [30000,32]x[32,1000] --------------------
    gat_ae_gemm_wmma<8><<<dim3(NNODES / 16, NPADD2 / 128), 32, 0, stream>>>(
        dbuf, FH, wt4, FH, FH, dec_b2, DIN, recon, DIN, 0);
}
